// RGCN_7456063226484
// MI455X (gfx1250) — matrix operand order repaired and checked by reading, not compiled, not run on hardware
//
#include <hip/hip_runtime.h>

// ---------------------------------------------------------------------------
// RGCN (2 layers) for MI455X / gfx1250, wave32.
//  - edge messages: relation-sorted 16-edge tiles, v_wmma_f32_16x16x32_bf16
//  - root GEMM   : full-precision v_wmma_f32_16x16x4_f32, fused scatter-mean
// ---------------------------------------------------------------------------

typedef __attribute__((ext_vector_type(16))) __bf16 bf16x16;
typedef __attribute__((ext_vector_type(8)))  float  f32x8;
typedef __attribute__((ext_vector_type(2)))  float  f32x2;

union BfVec { bf16x16 v; uint4 q[2]; };

__device__ __forceinline__ unsigned short f2bf(float f) {
    unsigned int u = __float_as_uint(f);
    unsigned int r = (u + 0x7FFFu + ((u >> 16) & 1u)) >> 16;   // RNE
    return (unsigned short)r;
}

// ---------------- sorting / degree kernels (run once) ----------------------

__global__ void k_hist(const int* __restrict__ edge_type,
                       const int* __restrict__ edge_index,
                       int* __restrict__ counts,
                       float* __restrict__ cnt, int E) {
    int e = blockIdx.x * blockDim.x + threadIdx.x;
    if (e >= E) return;
    atomicAdd(&counts[edge_type[e]], 1);
    atomicAdd(&cnt[edge_index[E + e]], 1.0f);   // in-degree at dst
}

__global__ void k_scan(const int* __restrict__ counts,
                       int* __restrict__ offsets,
                       int* __restrict__ cursor, int R) {
    if (blockIdx.x == 0 && threadIdx.x == 0) {
        int run = 0;
        for (int r = 0; r < R; ++r) {
            offsets[r] = run; cursor[r] = run; run += counts[r];
        }
        offsets[R] = run;
    }
}

__global__ void k_scatter(const int* __restrict__ edge_type,
                          int* __restrict__ cursor,
                          int* __restrict__ sorted, int E) {
    int e = blockIdx.x * blockDim.x + threadIdx.x;
    if (e >= E) return;
    int pos = atomicAdd(&cursor[edge_type[e]], 1);
    sorted[pos] = e;
}

// ---------------- per-layer kernels ----------------------------------------

// W[r][k][n] = sum_b att[r,b]*basis[b][k][n], stored bf16 pre-swizzled into
// this file's B-operand lane layout (contiguous K per lane half, same as A):
//   Wlay[(((r*2+kstep)*4+nt)*32 + lane)*16 + i]
//   lane<16 : column nt*16+lane,    K = kstep*32 + i        (i=0..15)
//   lane>=16: column nt*16+lane-16, K = kstep*32 + 16 + i
__global__ void k_build_w(const float* __restrict__ att,
                          const float* __restrict__ basis,
                          unsigned short* __restrict__ Wlay, int B) {
    int r = blockIdx.x;
    __shared__ float a_s[32];
    if (threadIdx.x < B) a_s[threadIdx.x] = att[r * B + threadIdx.x];
    __syncthreads();
    for (int e = threadIdx.x; e < 4096; e += blockDim.x) {
        int k = e >> 6, n = e & 63;
        float acc = 0.f;
        for (int b = 0; b < B; ++b) acc += a_s[b] * basis[b * 4096 + e];
        int kstep = k >> 5, krel = k & 31;
        int i = krel & 15, kh = krel >> 4;
        int nt = n >> 4, nl = n & 63 & 15;
        int lane = nl + (kh << 4);
        size_t idx = ((((size_t)r * 2 + kstep) * 4 + nt) * 32 + lane) * 16 + i;
        Wlay[idx] = f2bf(acc);
    }
}

// bf16 copy of x (optionally gathered through `gather` for layer 1)
__global__ void k_to_bf16(const float* __restrict__ x,
                          const int* __restrict__ gather,
                          unsigned short* __restrict__ out, int N) {
    int i = blockIdx.x * blockDim.x + threadIdx.x;
    if (i >= N * 64) return;
    float v;
    if (gather) { int n = i >> 6, k = i & 63; v = x[(size_t)gather[n] * 64 + k]; }
    else        { v = x[i]; }
    out[i] = f2bf(v);
}

// One wave per 16-edge tile of one relation: C(16x64) = A(16x64) @ W_r(64x64)
// via 8x v_wmma_f32_16x16x32_bf16, then scaled atomic scatter into s.
__global__ void k_edge_msg(const int* __restrict__ sorted,
                           const int* __restrict__ offsets,
                           const int* __restrict__ edge_index,
                           const float* __restrict__ edge_norm,
                           const unsigned short* __restrict__ Abf,
                           const unsigned short* __restrict__ Wlay,
                           float* __restrict__ s_out, int E) {
    const int r    = blockIdx.y;
    const int wave = threadIdx.x >> 5;
    const int lane = threadIdx.x & 31;
    const int off  = offsets[r];
    const int nEdg = offsets[r + 1] - off;
    const int nTil = (nEdg + 15) >> 4;
    const int wstr = gridDim.x * (blockDim.x >> 5);

    for (int tile = blockIdx.x * (blockDim.x >> 5) + wave; tile < nTil; tile += wstr) {
        const int j    = lane & 15;          // edge row within tile
        const int half = lane >> 4;
        const int slot = tile * 16 + j;
        const bool valid = slot < nEdg;
        const int eid  = valid ? sorted[off + slot] : 0;
        const int srcn = valid ? edge_index[eid]     : 0;
        const int dstn = valid ? edge_index[E + eid] : 0;
        const float nrm = valid ? edge_norm[eid] : 0.0f;

        // A operand: row j, contiguous K-half per lane half, as in Wlay (aligned 16B loads)
        const unsigned short* abase = Abf + (size_t)srcn * 64 + half * 16;
        BfVec a0, a1;
        a0.q[0] = *(const uint4*)(abase);        // K = ks*32 + {0..7|16..23}
        a0.q[1] = *(const uint4*)(abase + 8);    // K = ks*32 + {8..15|24..31}
        a1.q[0] = *(const uint4*)(abase + 32);
        a1.q[1] = *(const uint4*)(abase + 40);

        // broadcast per-output-row dst/norm (row m lives in lane m, m<16)
        const int mbase = half * 8;
        int dstR[8]; float nrmR[8];
        #pragma unroll
        for (int v = 0; v < 8; ++v) {
            dstR[v] = __shfl(dstn, mbase + v, 32);
            nrmR[v] = __shfl(nrm,  mbase + v, 32);
        }

        const unsigned short* wbase = Wlay + (size_t)r * 4096;
        #pragma unroll
        for (int nt = 0; nt < 4; ++nt) {
            const unsigned short* w0 = wbase + (size_t)((0 * 4 + nt) * 32 + lane) * 16;
            const unsigned short* w1 = wbase + (size_t)((1 * 4 + nt) * 32 + lane) * 16;
            BfVec b0, b1;
            b0.q[0] = *(const uint4*)(w0); b0.q[1] = *(const uint4*)(w0 + 8);
            b1.q[0] = *(const uint4*)(w1); b1.q[1] = *(const uint4*)(w1 + 8);

            f32x8 c = {0.f, 0.f, 0.f, 0.f, 0.f, 0.f, 0.f, 0.f};
            c = __builtin_amdgcn_wmma_f32_16x16x32_bf16(
                    false, a0.v, false, b0.v, (short)0, c, false, false);
            c = __builtin_amdgcn_wmma_f32_16x16x32_bf16(
                    false, a1.v, false, b1.v, (short)0, c, false, false);

            const int col = nt * 16 + j;
            #pragma unroll
            for (int v = 0; v < 8; ++v) {
                atomicAdd(&s_out[(size_t)dstR[v] * 64 + col], c[v] * nrmR[v]);
            }
        }
    }
}

// out[n] = s[n]/max(cnt,1) + x[n]@root + bias
// Root GEMM in full-precision f32 WMMA: one wave per 16-node tile,
// K=64 in 16 steps of v_wmma_f32_16x16x4_f32, 4 N-tiles accumulated.
// A wave reads/writes only its own 16 rows -> layer-2 in-place is safe.
__global__ void k_root_wmma(const float* __restrict__ s,
                            const float* __restrict__ cnt,
                            const float* __restrict__ x,
                            const int* __restrict__ gather,
                            const float* __restrict__ root,
                            const float* __restrict__ bias,
                            float* __restrict__ out, int N) {
    const int wave = threadIdx.x >> 5;
    const int lane = threadIdx.x & 31;
    const int mtile = blockIdx.x * (blockDim.x >> 5) + wave;
    const int nTiles = (N + 15) >> 4;
    if (mtile >= nTiles) return;           // uniform per wave: EXEC stays full

    const int j    = lane & 15;
    const int half = lane >> 4;
    const int node = mtile * 16 + j;
    const int nodeC = node < N ? node : N - 1;   // clamp for A loads
    const int row = gather ? gather[nodeC] : nodeC;
    const float* xr = x + (size_t)row * 64 + half * 2;   // K = ks*4 + half*2 + {0,1}

    f32x8 acc[4];
    #pragma unroll
    for (int nt = 0; nt < 4; ++nt)
        acc[nt] = (f32x8){0.f, 0.f, 0.f, 0.f, 0.f, 0.f, 0.f, 0.f};

    #pragma unroll
    for (int ks = 0; ks < 16; ++ks) {
        f32x2 a;
        a.x = xr[ks * 4];
        a.y = xr[ks * 4 + 1];
        const float* rb = root + (size_t)(ks * 4 + half * 2) * 64;
        #pragma unroll
        for (int nt = 0; nt < 4; ++nt) {
            const int col = nt * 16 + j;
            f32x2 b;
            b.x = rb[col];            // K row ks*4+half*2
            b.y = rb[64 + col];       // K row ks*4+half*2+1
            acc[nt] = __builtin_amdgcn_wmma_f32_16x16x4_f32(
                          false, a, false, b, (short)0, acc[nt], false, false);
        }
    }

    // epilogue: out = s/max(cnt,1) + acc + bias
    const int mb = half * 8;
    float cdiv[8];
    #pragma unroll
    for (int v = 0; v < 8; ++v) {
        const int nm = mtile * 16 + mb + v;
        float cc = (nm < N) ? cnt[nm] : 1.f;
        cdiv[v] = cc > 1.f ? cc : 1.f;
    }
    #pragma unroll
    for (int nt = 0; nt < 4; ++nt) {
        const int col = nt * 16 + j;
        const float bs = bias[col];
        #pragma unroll
        for (int v = 0; v < 8; ++v) {
            const int nm = mtile * 16 + mb + v;
            if (nm < N) {
                const size_t idx = (size_t)nm * 64 + col;
                out[idx] = s[idx] / cdiv[v] + acc[nt][v] + bs;
            }
        }
    }
}

// ---------------------------------------------------------------------------

extern "C" void kernel_launch(void* const* d_in, const int* in_sizes, int n_in,
                              void* d_out, int out_size, void* d_ws, size_t ws_size,
                              hipStream_t stream) {
    const int*   entity     = (const int*)d_in[0];
    const int*   edge_index = (const int*)d_in[1];
    const int*   edge_type  = (const int*)d_in[2];
    const float* edge_norm  = (const float*)d_in[3];
    const float* emb        = (const float*)d_in[4];
    const float* basis1     = (const float*)d_in[5];
    const float* att1       = (const float*)d_in[6];
    const float* root1      = (const float*)d_in[7];
    const float* bias1      = (const float*)d_in[8];
    const float* basis2     = (const float*)d_in[9];
    const float* att2       = (const float*)d_in[10];
    const float* root2      = (const float*)d_in[11];
    const float* bias2      = (const float*)d_in[12];

    const int N = in_sizes[0];
    const int E = in_sizes[2];
    const int B = in_sizes[5] / (64 * 64);
    const int R = in_sizes[6] / B;
    float* out = (float*)d_out;

    // workspace carve (256B aligned)
    char* p = (char*)d_ws;
    auto carve = [&](size_t bytes) -> void* {
        void* q = (void*)p; p += (bytes + 255) & ~(size_t)255; return q;
    };
    float*          s_buf   = (float*)carve((size_t)N * 64 * 4);
    float*          cnt     = (float*)carve((size_t)N * 4);
    unsigned short* Abf     = (unsigned short*)carve((size_t)N * 64 * 2);
    unsigned short* Wlay    = (unsigned short*)carve((size_t)R * 4096 * 2);
    int*            sorted  = (int*)carve((size_t)E * 4);
    int*            counts  = (int*)carve((size_t)R * 4);
    int*            offsets = (int*)carve((size_t)(R + 1) * 4);
    int*            cursor  = (int*)carve((size_t)R * 4);

    // --- sort edges by relation, compute in-degree (shared by both layers)
    hipMemsetAsync(counts, 0, (size_t)R * 4, stream);
    hipMemsetAsync(cnt,    0, (size_t)N * 4, stream);
    k_hist<<<(E + 255) / 256, 256, 0, stream>>>(edge_type, edge_index, counts, cnt, E);
    k_scan<<<1, 64, 0, stream>>>(counts, offsets, cursor, R);
    k_scatter<<<(E + 255) / 256, 256, 0, stream>>>(edge_type, cursor, sorted, E);

    const int elemBlocks = (N * 64 + 255) / 256;
    const dim3 egrid(32, R);
    const int nTiles = (N + 15) / 16;
    const int rootBlocks = (nTiles + 3) / 4;

    // --- layer 1: x = emb[entity]
    hipMemsetAsync(s_buf, 0, (size_t)N * 64 * 4, stream);
    k_build_w<<<R, 256, 0, stream>>>(att1, basis1, Wlay, B);
    k_to_bf16<<<elemBlocks, 256, 0, stream>>>(emb, entity, Abf, N);
    k_edge_msg<<<egrid, 128, 0, stream>>>(sorted, offsets, edge_index, edge_norm,
                                          Abf, Wlay, s_buf, E);
    k_root_wmma<<<rootBlocks, 128, 0, stream>>>(s_buf, cnt, emb, entity,
                                                root1, bias1, out, N);

    // --- layer 2: x = layer-1 output (d_out), result in-place to d_out
    hipMemsetAsync(s_buf, 0, (size_t)N * 64 * 4, stream);
    k_build_w<<<R, 256, 0, stream>>>(att2, basis2, Wlay, B);
    k_to_bf16<<<elemBlocks, 256, 0, stream>>>(out, nullptr, Abf, N);
    k_edge_msg<<<egrid, 128, 0, stream>>>(sorted, offsets, edge_index, edge_norm,
                                          Abf, Wlay, s_buf, E);
    k_root_wmma<<<rootBlocks, 128, 0, stream>>>(s_buf, cnt, out, nullptr,
                                                root2, bias2, out, N);
}
